// EggrollLinear_15058155339832
// MI455X (gfx1250) — matrix operand order repaired and checked by reading, not compiled, not run on hardware
//
#include <hip/hip_runtime.h>

#define OUT_N  4096
#define IN_K   4096
#define M_ROWS 8192   // 4 * 2048
#define GROUP  128

typedef __bf16 bf16_t;
typedef __attribute__((ext_vector_type(16))) bf16_t v16bf;
typedef __attribute__((ext_vector_type(8)))  float  v8f;

union FragBF {
    v16bf v;
    uint4 u[2];
};

// round-to-nearest-even f32 -> bf16, packed pair
__device__ __forceinline__ unsigned pack2bf(float lo, float hi) {
    unsigned ul = __builtin_bit_cast(unsigned, lo);
    unsigned uh = __builtin_bit_cast(unsigned, hi);
    unsigned rl = (ul + 0x7FFFu + ((ul >> 16) & 1u)) >> 16;
    unsigned rh = (uh + 0x7FFFu + ((uh >> 16) & 1u)) >> 16;
    return (rl & 0xFFFFu) | (rh << 16);
}

// 64-byte global -> LDS async copy (four b128 ops; IOFFSET applies to both
// the global and LDS addresses, so shared immediates cover both sides).
__device__ __forceinline__ void async_copy_64B(unsigned lds_off, const unsigned short* g) {
    asm volatile("global_load_async_to_lds_b128 %0, %1, off\n\t"
                 "global_load_async_to_lds_b128 %0, %1, off offset:16\n\t"
                 "global_load_async_to_lds_b128 %0, %1, off offset:32\n\t"
                 "global_load_async_to_lds_b128 %0, %1, off offset:48"
                 :: "v"(lds_off), "v"(g) : "memory");
}

__device__ __forceinline__ void wait_async0() {
    asm volatile("s_wait_asynccnt 0x0" ::: "memory");
}

// ---------------------------------------------------------------------------
// Kernel 0: x fp32 -> bf16 (one pass; keeps packing out of the GEMM hot loop)
// ---------------------------------------------------------------------------
__global__ __launch_bounds__(256)
void cvt_x_kernel(const float* __restrict__ x, unsigned short* __restrict__ xb) {
    const long long g    = (long long)blockIdx.x * blockDim.x + threadIdx.x;
    const long long base = g * 16;
    float4 f0 = *(const float4*)(x + base + 0);
    float4 f1 = *(const float4*)(x + base + 4);
    float4 f2 = *(const float4*)(x + base + 8);
    float4 f3 = *(const float4*)(x + base + 12);
    uint4 o0, o1;
    o0.x = pack2bf(f0.x, f0.y); o0.y = pack2bf(f0.z, f0.w);
    o0.z = pack2bf(f1.x, f1.y); o0.w = pack2bf(f1.z, f1.w);
    o1.x = pack2bf(f2.x, f2.y); o1.y = pack2bf(f2.z, f2.w);
    o1.z = pack2bf(f3.x, f3.y); o1.w = pack2bf(f3.z, f3.w);
    *(uint4*)(xb + base)     = o0;
    *(uint4*)(xb + base + 8) = o1;
}

// ---------------------------------------------------------------------------
// Kernel 1: dequantize ternary int8 * per-group scale -> bf16 weights.
// ---------------------------------------------------------------------------
__global__ __launch_bounds__(256)
void dequant_w_kernel(const signed char* __restrict__ tern,
                      const float* __restrict__ scales,
                      unsigned short* __restrict__ wb) {
    const long long g    = (long long)blockIdx.x * blockDim.x + threadIdx.x;
    const long long base = g * 16;
    const float s = scales[base >> 7];   // GROUP == 128

    union { uint4 u; signed char c[16]; } raw;
    raw.u = *(const uint4*)(tern + base);

    float f[16];
#pragma unroll
    for (int i = 0; i < 16; ++i) f[i] = (float)raw.c[i] * s;

    uint4 o0, o1;
    o0.x = pack2bf(f[0],  f[1]);  o0.y = pack2bf(f[2],  f[3]);
    o0.z = pack2bf(f[4],  f[5]);  o0.w = pack2bf(f[6],  f[7]);
    o1.x = pack2bf(f[8],  f[9]);  o1.y = pack2bf(f[10], f[11]);
    o1.z = pack2bf(f[12], f[13]); o1.w = pack2bf(f[14], f[15]);

    *(uint4*)(wb + base)     = o0;
    *(uint4*)(wb + base + 8) = o1;
}

// ---------------------------------------------------------------------------
// Kernel 2: t[m] = sigma * dot(x[m,:], b)   (one block per row)
// ---------------------------------------------------------------------------
__global__ __launch_bounds__(256)
void row_dot_b_kernel(const float* __restrict__ x,
                      const float* __restrict__ bvec,
                      const float* __restrict__ sigma,
                      float* __restrict__ t) {
    __shared__ float red[256];
    const int m = blockIdx.x;
    const float* xr = x + (size_t)m * IN_K;
    float s = 0.f;
    for (int k = threadIdx.x; k < IN_K; k += 256) s += xr[k] * bvec[k];
    red[threadIdx.x] = s;
    __syncthreads();
#pragma unroll
    for (int off = 128; off > 0; off >>= 1) {
        if (threadIdx.x < off) red[threadIdx.x] += red[threadIdx.x + off];
        __syncthreads();
    }
    if (threadIdx.x == 0) t[m] = red[0] * sigma[0];
}

// ---------------------------------------------------------------------------
// Kernel 3: 128x128 bf16 WMMA GEMM; K=64 per double-buffered LDS stage
// (16 WMMAs per barrier), async-to-LDS DMA staging, fused rank-1 epilogue.
// 8 waves: 4(M) x 2(N); each wave computes 32x64 = 2x4 WMMA accumulators.
// ---------------------------------------------------------------------------
__global__ __launch_bounds__(256, 1)
void gemm_bf16_wmma_kernel(const unsigned short* __restrict__ xb,
                           const unsigned short* __restrict__ wb,
                           const float* __restrict__ avec,
                           const float* __restrict__ t,
                           float* __restrict__ out) {
    // 64 halves of K per stage, padded to 72 (144 B stride, 16B-aligned)
    __shared__ __align__(16) unsigned short As[2][128][72];
    __shared__ __align__(16) unsigned short Bs[2][128][72];

    const int tid  = threadIdx.x;
    const int lane = tid & 31;
    const int wave = tid >> 5;      // 0..7
    const int wm   = wave >> 1;     // 0..3  (M)
    const int wn   = wave & 1;      // 0..1  (N)
    const int hi   = lane >> 4;
    const int ln   = lane & 15;

    const int Nbase = blockIdx.x * 128;
    const int Mbase = blockIdx.y * 128;

    // staging coords: 2 threads per row, 32 bf16 (64 B) each
    const int lr = tid >> 1;            // 0..127
    const int lc = (tid & 1) * 32;      // 0 or 32

    const unsigned short* xrow = xb + (size_t)(Mbase + lr) * IN_K + lc;
    const unsigned short* wrow = wb + (size_t)(Nbase + lr) * IN_K + lc;

    // LDS byte offsets for this thread's async destinations (both buffers)
    const unsigned asBase = (unsigned)(unsigned long long)&As[0][0][0];
    const unsigned bsBase = (unsigned)(unsigned long long)&Bs[0][0][0];
    const unsigned rowOff = (unsigned)(lr * 72 + lc) * 2u;
    const unsigned bufStride = 128u * 72u * 2u;
    const unsigned aDst0 = asBase + rowOff, aDst1 = asBase + bufStride + rowOff;
    const unsigned bDst0 = bsBase + rowOff, bDst1 = bsBase + bufStride + rowOff;

    v8f acc[2][4];
    {
        v8f zero = {};
#pragma unroll
        for (int i = 0; i < 2; ++i)
#pragma unroll
            for (int j = 0; j < 4; ++j) acc[i][j] = zero;
    }

    const int NK = IN_K / 64;   // 64 stages

    // ---- prologue: async-stage k-tile 0 into buffer 0 ----
    async_copy_64B(aDst0, xrow);
    async_copy_64B(bDst0, wrow);
    wait_async0();
    __syncthreads();

    for (int kt = 0; kt < NK; ++kt) {
        const int buf = kt & 1;
        const bool have_next = (kt + 1) < NK;

        // ---- issue async DMA for next stage into the other buffer ----
        if (have_next) {
            const unsigned short* xg = xrow + (kt + 1) * 64;
            const unsigned short* wg = wrow + (kt + 1) * 64;
            async_copy_64B(buf ? aDst0 : aDst1, xg);
            async_copy_64B(buf ? bDst0 : bDst1, wg);
        }

        // ---- two K=32 sub-steps per stage: 16 WMMAs per barrier ----
#pragma unroll
        for (int kk = 0; kk < 2; ++kk) {
            FragBF fa[2], fb[4];
#pragma unroll
            for (int ms = 0; ms < 2; ++ms) {
                const unsigned short* ap =
                    &As[buf][wm * 32 + ms * 16 + ln][kk * 32 + hi * 8];
                fa[ms].u[0] = *(const uint4*)(ap);
                fa[ms].u[1] = *(const uint4*)(ap + 16);
            }
#pragma unroll
            for (int ns = 0; ns < 4; ++ns) {
                const unsigned short* bp =
                    &Bs[buf][wn * 64 + ns * 16 + ln][kk * 32 + hi * 8];
                fb[ns].u[0] = *(const uint4*)(bp);
                fb[ns].u[1] = *(const uint4*)(bp + 16);
            }
#pragma unroll
            for (int ms = 0; ms < 2; ++ms)
#pragma unroll
                for (int ns = 0; ns < 4; ++ns)
                    acc[ms][ns] = __builtin_amdgcn_wmma_f32_16x16x32_bf16(
                        false, fa[ms].v, false, fb[ns].v,
                        (short)0, acc[ms][ns], false, false);
        }

        // ---- DMA complete + make visible to all waves ----
        if (have_next) wait_async0();
        __syncthreads();
    }

    // ---- epilogue: y += t[m] * a[n], store fp32 ----
    float aval[4];
#pragma unroll
    for (int ns = 0; ns < 4; ++ns)
        aval[ns] = avec[Nbase + wn * 64 + ns * 16 + ln];

    float tval[2][8];
#pragma unroll
    for (int ms = 0; ms < 2; ++ms)
#pragma unroll
        for (int v = 0; v < 8; ++v)
            tval[ms][v] = t[Mbase + wm * 32 + ms * 16 + hi * 8 + v];

#pragma unroll
    for (int ms = 0; ms < 2; ++ms) {
#pragma unroll
        for (int ns = 0; ns < 4; ++ns) {
#pragma unroll
            for (int v = 0; v < 8; ++v) {
                const int m = Mbase + wm * 32 + ms * 16 + hi * 8 + v;
                const int n = Nbase + wn * 64 + ns * 16 + ln;
                out[(size_t)m * OUT_N + n] = acc[ms][ns][v] + tval[ms][v] * aval[ns];
            }
        }
    }
}

// ---------------------------------------------------------------------------
extern "C" void kernel_launch(void* const* d_in, const int* in_sizes, int n_in,
                              void* d_out, int out_size, void* d_ws, size_t ws_size,
                              hipStream_t stream) {
    const float*       x      = (const float*)d_in[0];
    const float*       a      = (const float*)d_in[1];
    const float*       b      = (const float*)d_in[2];
    const float*       sigma  = (const float*)d_in[3];
    const signed char* tern   = (const signed char*)d_in[4];
    const float*       scales = (const float*)d_in[5];
    float*             out    = (float*)d_out;

    // ws layout: [bf16 W][bf16 x][t]
    unsigned short* wb   = (unsigned short*)d_ws;
    unsigned short* xbuf = (unsigned short*)((char*)d_ws + (size_t)OUT_N * IN_K * 2);
    float* t = (float*)((char*)d_ws + (size_t)OUT_N * IN_K * 2 + (size_t)M_ROWS * IN_K * 2);

    // 0) x -> bf16 (once; reused by 32 N-blocks)
    cvt_x_kernel<<<(M_ROWS * (IN_K / 16)) / 256, 256, 0, stream>>>(x, xbuf);

    // 1) dequantize weights to bf16 (once; reused by 64 M-blocks)
    dequant_w_kernel<<<(OUT_N * (IN_K / 16)) / 256, 256, 0, stream>>>(tern, scales, wb);

    // 2) per-row correction scalar t[m] = sigma * (x[m,:] . b)
    row_dot_b_kernel<<<M_ROWS, 256, 0, stream>>>(x, b, sigma, t);

    // 3) fused WMMA GEMM + rank-1 correction
    dim3 grid(OUT_N / 128, M_ROWS / 128);   // (32, 64)
    gemm_bf16_wmma_kernel<<<grid, 256, 0, stream>>>(xbuf, wb, a, t, out);
}
